// ImageEncoder_25838523253482
// MI455X (gfx1250) — matrix operand order repaired and checked by reading, not compiled, not run on hardware
//
#include <hip/hip_runtime.h>

// ---------------------------------------------------------------------------
// Types for CDNA5 WMMA (gfx1250, wave32)
// ---------------------------------------------------------------------------
typedef __attribute__((ext_vector_type(16))) __bf16 v16bf;
typedef __attribute__((ext_vector_type(8)))  float  v8f;

union ABfrag { v16bf v; uint4 q[2]; };   // 32 bytes: 16 bf16
union F8     { v8f   v; float f[8]; };

__device__ __forceinline__ v8f wmma_bf16(v16bf a, v16bf b, v8f c) {
  // D = A(16x32) * B(32x16) + C(16x16), f32 accum
  return __builtin_amdgcn_wmma_f32_16x16x32_bf16(
      /*neg_a=*/false, a, /*neg_b=*/false, b,
      /*c_mod=*/(short)0, c, /*reuse_a=*/false, /*reuse_b=*/false);
}

// ---------------------------------------------------------------------------
// Problem constants
// ---------------------------------------------------------------------------
#define BATCH   16
#define HIDDEN  768
#define VOCAB   8192
#define NTOK    (BATCH * 32 * 32)        // 16384 feature rows

// ---------------------------------------------------------------------------
// Kernel: repack conv weights OIHW (o, ci, kh, kw) fp32 -> [o][pos][ci] bf16
// so that K ordering k = pos*CIN + ci gives contiguous A and B fragments.
// ---------------------------------------------------------------------------
__global__ void repack_w_kernel(const float* __restrict__ w,
                                __bf16* __restrict__ wp,
                                int cout, int cin) {
  int t = blockIdx.x * blockDim.x + threadIdx.x;
  int tot = cout * cin * 9;
  if (t >= tot) return;
  int o   = t / (cin * 9);
  int r   = t - o * (cin * 9);
  int ci  = r / 9;
  int pos = r - ci * 9;
  wp[((long long)o * 9 + pos) * cin + ci] = (__bf16)w[t];
}

// ---------------------------------------------------------------------------
// Kernel: codebook -> bf16 copy + per-row squared norm (fp32)
// ---------------------------------------------------------------------------
__global__ __launch_bounds__(256)
void cb_prep_kernel(const float* __restrict__ cb,
                    __bf16* __restrict__ cbb,
                    float* __restrict__ c2) {
  int v = blockIdx.x;                       // 0..VOCAB-1
  const float* row = cb + (long long)v * HIDDEN;
  __bf16* orow = cbb + (long long)v * HIDDEN;
  float s = 0.f;
  for (int j = threadIdx.x; j < HIDDEN; j += 256) {
    float f = row[j];
    orow[j] = (__bf16)f;
    s += f * f;
  }
#pragma unroll
  for (int off = 16; off > 0; off >>= 1) s += __shfl_xor(s, off, 32);
  __shared__ float red[8];
  if ((threadIdx.x & 31) == 0) red[threadIdx.x >> 5] = s;
  __syncthreads();
  if (threadIdx.x == 0) {
    float tot = 0.f;
#pragma unroll
    for (int i = 0; i < 8; i++) tot += red[i];
    c2[v] = tot;
  }
}

// ---------------------------------------------------------------------------
// Kernel: conv1 (3 -> 192, stride 2, pad 1) direct; K=27 is too small for
// WMMA. Input NCHW fp32, output NHWC bf16 + ReLU. Output offset == tid.
// ---------------------------------------------------------------------------
__global__ __launch_bounds__(256)
void conv1_direct_kernel(const float* __restrict__ x,   // [16,3,256,256]
                         const float* __restrict__ w,   // [192,3,3,3]
                         const float* __restrict__ bias,
                         __bf16* __restrict__ out) {    // [16,128,128,192]
  const int HIN = 256, HOUT = 128, COUT = 192;
  long long t = (long long)blockIdx.x * blockDim.x + threadIdx.x;
  const long long TOT = (long long)BATCH * HOUT * HOUT * COUT;
  if (t >= TOT) return;
  int oc = (int)(t % COUT);
  long long pix = t / COUT;
  int b  = (int)(pix / (HOUT * HOUT));
  int pr = (int)(pix - (long long)b * HOUT * HOUT);
  int oh = pr / HOUT, ow = pr - (pr / HOUT) * HOUT;
  float s = bias[oc];
  const float* wk = w + oc * 27;
#pragma unroll
  for (int ci = 0; ci < 3; ci++) {
#pragma unroll
    for (int kh = 0; kh < 3; kh++) {
      int iy = 2 * oh - 1 + kh;
      if ((unsigned)iy >= (unsigned)HIN) continue;
#pragma unroll
      for (int kw = 0; kw < 3; kw++) {
        int ix = 2 * ow - 1 + kw;
        if ((unsigned)ix >= (unsigned)HIN) continue;
        s += x[(((long long)b * 3 + ci) * HIN + iy) * HIN + ix] *
             wk[ci * 9 + kh * 3 + kw];
      }
    }
  }
  s = fmaxf(s, 0.f);
  out[t] = (__bf16)s;
}

// ---------------------------------------------------------------------------
// Kernel: implicit-GEMM conv with WMMA bf16, wave tile 32(M) x 64(N).
//   M = B*HOUT*HOUT (rows = output pixels), N = COUT, K = 9*CIN
//   K order: k = (kh*3+kw)*CIN + ci  -> each 32-chunk inside one (kh,kw)
//   in : NHWC bf16 [B,HIN,HIN,CIN];  wp : [COUT][9][CIN] bf16
//   out: NHWC bf16 [B,HOUT,HOUT,COUT]  (offset = m*COUT + n)
// Block: 256 threads = 8 waves, arranged 4(M) x 2(N); block tile 128 x 128.
// K stepped by 64 with ping-pong A/B fragment buffers: loads for the next
// 32-slice are issued before the WMMAs of the current slice (latency hiding).
// ---------------------------------------------------------------------------
template <int CIN, int COUT, int HIN, int HOUT, bool RELU>
__global__ __launch_bounds__(256)
void conv_wmma_kernel(const __bf16* __restrict__ in,
                      const __bf16* __restrict__ wp,
                      const float*  __restrict__ bias,
                      __bf16* __restrict__ out) {
  constexpr int K   = 9 * CIN;              // K/64 is integral (27 or 54)
  constexpr int PIX = HOUT * HOUT;
  const int lane = threadIdx.x & 31;
  const int wave = threadIdx.x >> 5;
  const int wm = wave & 3, wn = wave >> 2;
  const int mblk = blockIdx.x * 128 + wm * 32;
  const int nblk = blockIdx.y * 128 + wn * 64;
  const int lrow = lane & 15;
  const int half = lane >> 4;

  // This lane's two A rows (16-bit A 16x32: lanes 0-15 and 16-31 both M=0-15)
  int bb_[2], oh_[2], ow_[2];
#pragma unroll
  for (int mi = 0; mi < 2; mi++) {
    const int m  = mblk + mi * 16 + lrow;
    const int b  = m / PIX;
    const int pr = m - b * PIX;
    bb_[mi] = b;
    oh_[mi] = pr / HOUT;
    ow_[mi] = pr - (pr / HOUT) * HOUT;
  }

  // Per-tile B row pointers (loop-invariant -> K offsets fold to immediates)
  const __bf16* wt[4];
#pragma unroll
  for (int t = 0; t < 4; t++)
    wt[t] = wp + (long long)(nblk + t * 16 + lrow) * K + half * 16;

  auto loadB4 = [&](int kk, ABfrag* bf) {
#pragma unroll
    for (int t = 0; t < 4; t++) {
      bf[t].q[0] = *(const uint4*)(wt[t] + kk);
      bf[t].q[1] = *(const uint4*)(wt[t] + kk + 8);
    }
  };
  auto loadA2 = [&](int kk, ABfrag* a) {
    const int pos = kk / CIN;          // CIN % 32 == 0: chunk within one pos
    const int ci0 = kk - pos * CIN;
    const int kh = pos / 3, kw = pos - (pos / 3) * 3;
#pragma unroll
    for (int mi = 0; mi < 2; mi++) {
      const int iy = 2 * oh_[mi] - 1 + kh;
      const int ix = 2 * ow_[mi] - 1 + kw;
      if ((unsigned)iy < (unsigned)HIN && (unsigned)ix < (unsigned)HIN) {
        const __bf16* src = in +
            ((((long long)bb_[mi] * HIN + iy) * HIN + ix) * CIN + ci0) + half * 16;
        a[mi].q[0] = *(const uint4*)src;
        a[mi].q[1] = *(const uint4*)(src + 8);
      } else {
        a[mi].q[0] = make_uint4(0, 0, 0, 0);
        a[mi].q[1] = make_uint4(0, 0, 0, 0);
      }
    }
  };

  F8 acc[2][4];
#pragma unroll
  for (int mi = 0; mi < 2; mi++)
#pragma unroll
    for (int t = 0; t < 4; t++) acc[mi][t].v = (v8f){0, 0, 0, 0, 0, 0, 0, 0};

  ABfrag b0[4], b1[4], a0[2], a1[2];
  loadB4(0, b0);
  loadA2(0, a0);

  for (int kk = 0; kk < K; kk += 64) {
    // stage 1: prefetch kk+32, compute kk
    loadB4(kk + 32, b1);
    loadA2(kk + 32, a1);
#pragma unroll
    for (int t = 0; t < 4; t++) {
      acc[0][t].v = wmma_bf16(a0[0].v, b0[t].v, acc[0][t].v);
      acc[1][t].v = wmma_bf16(a0[1].v, b0[t].v, acc[1][t].v);
    }
    // stage 2: prefetch kk+64, compute kk+32
    if (kk + 64 < K) {
      loadB4(kk + 64, b0);
      loadA2(kk + 64, a0);
    }
#pragma unroll
    for (int t = 0; t < 4; t++) {
      acc[0][t].v = wmma_bf16(a1[0].v, b1[t].v, acc[0][t].v);
      acc[1][t].v = wmma_bf16(a1[1].v, b1[t].v, acc[1][t].v);
    }
  }

  // Epilogue: D elem r -> row (mblk + mi*16 + r + 8*half), col (nblk+t*16+lrow)
#pragma unroll
  for (int t = 0; t < 4; t++) {
    const int n  = nblk + t * 16 + lrow;
    const float bv = bias[n];
#pragma unroll
    for (int mi = 0; mi < 2; mi++) {
#pragma unroll
      for (int r = 0; r < 8; r++) {
        float vv = acc[mi][t].f[r] + bv;
        if (RELU) vv = fmaxf(vv, 0.f);
        out[(long long)(mblk + mi * 16 + r + 8 * half) * COUT + n] = (__bf16)vv;
      }
    }
  }
}

// ---------------------------------------------------------------------------
// Kernel: fused VQ GEMM + argmin.
//   feats [NTOK,768] bf16, cbb [VOCAB,768] bf16 (L2-resident), c2 [VOCAB] f32
//   score(n) = ||c_n||^2 - 2 * f.c_n   (||f||^2 is argmin-invariant)
// Block = 64 threads (2 waves). 32-row A panel in LDS, staged with the CDNA5
// async copy path (global_load_async_to_lds_b128 + s_wait_asynccnt) and
// SHARED by both waves; waves split the vocab into interleaved 64-wide tiles.
// B fragments are ping-pong double-buffered over K (step 64); the final stage
// of each vocab tile loads the FIRST K-slice of the next tile, so the argmin
// epilogue overlaps the next tile's loads.
// Row pad 776 -> row stride 1552B = 388 dwords, 388 % 64 == 4: conflict-free.
// ---------------------------------------------------------------------------
__global__ __launch_bounds__(64)
void vq_argmin_kernel(const __bf16* __restrict__ feats,
                      const __bf16* __restrict__ cbb,
                      const float*  __restrict__ c2,
                      int* __restrict__ tokens) {
  __shared__ __bf16 apan[32][776];           // 49,664 bytes
  __shared__ float  mrg_v[2][32];
  __shared__ int    mrg_i[2][32];
  const int lane = threadIdx.x & 31;
  const int wave = threadIdx.x >> 5;         // 0..1
  const int lrow = lane & 15;
  const int half = lane >> 4;
  const int mblk = blockIdx.x * 32;

  // ---- Stage 32x768 A panel: async global->LDS (GVS mode), b128 chunks ----
  {
    const int row = lane;                    // each lane owns one panel row
    const int hsel = wave;                   // wave w copies columns w*384..
    unsigned ldsoff = (unsigned)(uintptr_t)(&apan[row][hsel * 384]);
    unsigned voff   = (unsigned)(((unsigned)(mblk + row) * HIDDEN + hsel * 384) * 2);
    unsigned long long gbase = (unsigned long long)(uintptr_t)feats;
#pragma unroll 4
    for (int j = 0; j < 48; ++j) {           // 48 * 16B = 768B per lane
      asm volatile("global_load_async_to_lds_b128 %0, %1, %2"
                   :
                   : "v"(ldsoff), "v"(voff), "s"(gbase)
                   : "memory");
      ldsoff += 16;
      voff   += 16;
    }
    asm volatile("s_wait_asynccnt 0" ::: "memory");
  }
  __syncthreads();

  auto loadB4 = [&](int nrow0, int kk, ABfrag* bf) {
#pragma unroll
    for (int t = 0; t < 4; t++) {
      const __bf16* crow =
          cbb + (long long)(nrow0 + t * 16 + lrow) * HIDDEN + kk + half * 16;
      bf[t].q[0] = *(const uint4*)crow;
      bf[t].q[1] = *(const uint4*)(crow + 8);
    }
  };
  auto loadA2 = [&](int kk, ABfrag* a) {
#pragma unroll
    for (int mi = 0; mi < 2; mi++) {
      a[mi].q[0] = *(const uint4*)&apan[mi * 16 + lrow][kk + half * 16];
      a[mi].q[1] = *(const uint4*)&apan[mi * 16 + lrow][kk + 8 + half * 16];
    }
  };

  float best[2][8];
  int   bidx[2][8];
#pragma unroll
  for (int mi = 0; mi < 2; mi++)
#pragma unroll
    for (int r = 0; r < 8; r++) { best[mi][r] = 3.402823466e38f; bidx[mi][r] = 0; }

  ABfrag b0[4], b1[4];
  loadB4(wave * 64, 0, b0);                  // prime the pipeline

  // Waves sweep interleaved 64-wide vocab tiles: wave w -> n0 = w*64 + 128*i
  for (int n0 = wave * 64; n0 < VOCAB; n0 += 128) {
    const int nnext = (n0 + 128 < VOCAB) ? (n0 + 128) : wave * 64;

    F8 acc[2][4];
#pragma unroll
    for (int mi = 0; mi < 2; mi++)
#pragma unroll
      for (int t = 0; t < 4; t++) acc[mi][t].v = (v8f){0, 0, 0, 0, 0, 0, 0, 0};

    for (int kk = 0; kk < HIDDEN; kk += 64) {
      // stage 1: prefetch kk+32, compute kk with b0
      loadB4(n0, kk + 32, b1);
      ABfrag a[2];
      loadA2(kk, a);
#pragma unroll
      for (int t = 0; t < 4; t++) {
        acc[0][t].v = wmma_bf16(a[0].v, b0[t].v, acc[0][t].v);
        acc[1][t].v = wmma_bf16(a[1].v, b0[t].v, acc[1][t].v);
      }
      // stage 2: prefetch kk+64 (or next tile's first slice), compute kk+32
      if (kk + 64 < HIDDEN) loadB4(n0, kk + 64, b0);
      else                  loadB4(nnext, 0, b0);
      loadA2(kk + 32, a);
#pragma unroll
      for (int t = 0; t < 4; t++) {
        acc[0][t].v = wmma_bf16(a[0].v, b1[t].v, acc[0][t].v);
        acc[1][t].v = wmma_bf16(a[1].v, b1[t].v, acc[1][t].v);
      }
    }

    // scores + running argmin (tie -> lower index, matches jnp.argmin);
    // this ALU block overlaps the in-flight loads for the next vocab tile.
#pragma unroll
    for (int t = 0; t < 4; t++) {
      const int n = n0 + t * 16 + lrow;
      const float cn = c2[n];
#pragma unroll
      for (int mi = 0; mi < 2; mi++) {
#pragma unroll
        for (int r = 0; r < 8; r++) {
          float sc = cn - 2.f * acc[mi][t].f[r];
          if (sc < best[mi][r] || (sc == best[mi][r] && n < bidx[mi][r])) {
            best[mi][r] = sc;
            bidx[mi][r] = n;
          }
        }
      }
    }
  }

  // Reduce across the 16 lanes that share each D row (masks stay in-half)
#pragma unroll
  for (int off = 8; off > 0; off >>= 1) {
#pragma unroll
    for (int mi = 0; mi < 2; mi++) {
#pragma unroll
      for (int r = 0; r < 8; r++) {
        float ov = __shfl_xor(best[mi][r], off, 32);
        int   oi = __shfl_xor(bidx[mi][r], off, 32);
        if (ov < best[mi][r] || (ov == best[mi][r] && oi < bidx[mi][r])) {
          best[mi][r] = ov;
          bidx[mi][r] = oi;
        }
      }
    }
  }
  if (lrow == 0) {
#pragma unroll
    for (int mi = 0; mi < 2; mi++)
#pragma unroll
      for (int r = 0; r < 8; r++) {
        mrg_v[wave][mi * 16 + r + 8 * half] = best[mi][r];
        mrg_i[wave][mi * 16 + r + 8 * half] = bidx[mi][r];
      }
  }
  __syncthreads();

  // Cross-wave merge: first 32 threads pick the winner per row
  if (threadIdx.x < 32) {
    const int row = threadIdx.x;
    float v0 = mrg_v[0][row], v1 = mrg_v[1][row];
    int   i0 = mrg_i[0][row], i1 = mrg_i[1][row];
    int win = (v1 < v0 || (v1 == v0 && i1 < i0)) ? i1 : i0;
    tokens[mblk + row] = win;
  }
}

// ---------------------------------------------------------------------------
// Kernel: write tokens (as float) + gather embedding rows into d_out.
// d_out layout: [0,16384) tokens, then [16384, 16384+16384*768) embeddings.
// ---------------------------------------------------------------------------
__global__ __launch_bounds__(192)
void gather_kernel(const int* __restrict__ tokens,
                   const float* __restrict__ emb,   // [VOCAB,768]
                   float* __restrict__ out) {
  const int m = blockIdx.x;                 // 0..NTOK-1
  const int tok = tokens[m];
  if (threadIdx.x == 0) out[m] = (float)tok;
  float* dst = out + NTOK + (long long)m * HIDDEN;
  const float* src = emb + (long long)tok * HIDDEN;
  const int j = threadIdx.x * 4;            // 192 threads * 4 = 768
  *(float4*)(dst + j) = *(const float4*)(src + j);
}

// ---------------------------------------------------------------------------
// Host side
// ---------------------------------------------------------------------------
extern "C" void kernel_launch(void* const* d_in, const int* in_sizes, int n_in,
                              void* d_out, int out_size, void* d_ws,
                              size_t ws_size, hipStream_t stream) {
  (void)in_sizes; (void)n_in; (void)out_size; (void)ws_size;

  const float* image = (const float*)d_in[0];   // [16,3,256,256]
  const float* w1 = (const float*)d_in[1];      // [192,3,3,3]
  const float* b1 = (const float*)d_in[2];
  const float* w2 = (const float*)d_in[3];      // [384,192,3,3]
  const float* b2 = (const float*)d_in[4];
  const float* w3 = (const float*)d_in[5];      // [768,384,3,3]
  const float* b3 = (const float*)d_in[6];
  const float* cb  = (const float*)d_in[7];     // [8192,768]
  const float* emb = (const float*)d_in[8];     // [8192,768]
  float* out = (float*)d_out;

  // Workspace carve-up (all bf16 activations, NHWC)
  char* ws = (char*)d_ws;
  size_t off = 0;
  auto carve = [&](size_t bytes) -> char* {
    char* p = ws + off;
    off = (off + bytes + 255) & ~(size_t)255;
    return p;
  };
  __bf16* f1    = (__bf16*)carve((size_t)16 * 128 * 128 * 192 * 2); // 100.7 MB
  __bf16* f2    = (__bf16*)carve((size_t)16 * 64 * 64 * 384 * 2);   //  50.3 MB
  __bf16* feats = (__bf16*)carve((size_t)NTOK * HIDDEN * 2);        //  25.2 MB
  __bf16* wp2   = (__bf16*)carve((size_t)384 * 1728 * 2);
  __bf16* wp3   = (__bf16*)carve((size_t)768 * 3456 * 2);
  __bf16* cbb   = (__bf16*)carve((size_t)VOCAB * HIDDEN * 2);       //  12.6 MB
  float*  c2v   = (float*)carve((size_t)VOCAB * 4);
  int*    tok   = (int*)carve((size_t)NTOK * 4);

  // Prep: weight repack + codebook bf16/norms
  repack_w_kernel<<<(384 * 192 * 9 + 255) / 256, 256, 0, stream>>>(w2, wp2, 384, 192);
  repack_w_kernel<<<(768 * 384 * 9 + 255) / 256, 256, 0, stream>>>(w3, wp3, 768, 384);
  cb_prep_kernel<<<VOCAB, 256, 0, stream>>>(cb, cbb, c2v);

  // conv1: direct (K=27), output NHWC bf16 + ReLU
  {
    long long tot = (long long)16 * 128 * 128 * 192;
    conv1_direct_kernel<<<(unsigned)(tot / 256), 256, 0, stream>>>(image, w1, b1, f1);
  }

  // conv2: 192 -> 384, 128 -> 64, ReLU.  M=65536, N=384, K=1728
  conv_wmma_kernel<192, 384, 128, 64, true>
      <<<dim3(65536 / 128, 384 / 128), 256, 0, stream>>>(f1, wp2, b2, f2);

  // conv3: 384 -> 768, 64 -> 32, no ReLU.  M=16384, N=768, K=3456
  conv_wmma_kernel<384, 768, 64, 32, false>
      <<<dim3(16384 / 128, 768 / 128), 256, 0, stream>>>(f2, wp3, b3, feats);

  // VQ: fused GEMM + argmin over vocab
  vq_argmin_kernel<<<NTOK / 32, 64, 0, stream>>>(feats, cbb, c2v, tok);

  // tokens + embedding gather into d_out
  gather_kernel<<<NTOK, 192, 0, stream>>>(tok, emb, out);
}
